// JAM_TFR_20177756357081
// MI455X (gfx1250) — compile-verified, matrix operand order checked by reading, not run on hardware
//
#include <hip/hip_runtime.h>

// ---------------------------------------------------------------------------
// CDNA5 (gfx1250) implementation: bf16 WMMA for all GEMM-shaped work
// (convs via implicit GEMM, LSTM recurrence, attention projections, MLP head),
// VALU for elementwise / softmax / batchnorm glue.
// GEMM/conv blocks: 128x64 tile, 8 waves, each wave owns a 32x32 quadrant
// (2 A-frags + 2 B-frags -> 4 WMMAs) to halve LDS traffic per FLOP.
// ---------------------------------------------------------------------------

typedef __attribute__((ext_vector_type(16))) __bf16 v16bf;
typedef __attribute__((ext_vector_type(8)))  float  v8f;

#define BDIM 256

// ---- scalar bf16 helpers ---------------------------------------------------
__device__ __forceinline__ unsigned short f2bf(float f) {
  union { float f; unsigned int u; } x; x.f = f;
  unsigned int u = x.u;
  unsigned int r = (u + 0x7FFFu + ((u >> 16) & 1u)) >> 16;
  return (unsigned short)r;
}
__device__ __forceinline__ float bf2f(unsigned short h) {
  union { unsigned int u; float f; } x; x.u = ((unsigned int)h) << 16;
  return x.f;
}
__device__ __forceinline__ float sigmf(float x) { return 1.0f / (1.0f + expf(-x)); }

// ---- WMMA helpers ----------------------------------------------------------
__device__ __forceinline__ v8f wmma_bf16(v16bf a, v16bf b, v8f c) {
  // D = A(16x32) * B(32x16) + C ; f32 accumulate
  return __builtin_amdgcn_wmma_f32_16x16x32_bf16(false, a, false, b, (short)0, c,
                                                 false, false);
}

// Load a 16x32 bf16 fragment from LDS laid out row-major [16][rstride].
// Same routine serves the A operand (row = M) and the transposed-weight B
// operand (row = N, since B[k][n] = W[n][k]).
// ISA layout: lanes 0-15 row r=lane, K = {0..7,16..23}; lanes 16-31 K += 8.
__device__ __forceinline__ v16bf frag_ld(const unsigned short* base, int rstride,
                                         int kOff) {
  int lane = threadIdx.x & 31;
  int r = lane & 15;
  int kb = kOff + ((lane & 16) ? 8 : 0);
  union { unsigned int u[8]; v16bf v; } f;
#pragma unroll
  for (int i = 0; i < 8; ++i) {
    int k = kb + ((i < 4) ? (i * 2) : (16 + (i - 4) * 2));
    f.u[i] = *(const unsigned int*)(base + r * rstride + k);
  }
  return f.v;
}

// Store one 16x16 f32 D tile (8 VGPRs/lane layout) with bias / relu / masks.
__device__ __forceinline__ void store_tile(float* C, int ldc, int m0, int n0,
                                           int M, int N, v8f acc,
                                           const float* bias, int relu) {
  int lane = threadIdx.x & 31;
  int n = n0 + (lane & 15);
  if (n >= N) return;
  int mb = m0 + ((lane & 16) ? 8 : 0);
  float bv = bias ? bias[n] : 0.0f;
#pragma unroll
  for (int j = 0; j < 8; ++j) {
    int m = mb + j;
    if (m < M) {
      float v = acc[j] + bv;
      if (relu) v = fmaxf(v, 0.0f);
      C[(size_t)m * ldc + n] = v;
    }
  }
}

// ---------------------------------------------------------------------------
// Elementwise utility kernels
// ---------------------------------------------------------------------------
__global__ void f2bf_kernel(const float* __restrict__ in,
                            unsigned short* __restrict__ out, int n) {
  int i = blockIdx.x * blockDim.x + threadIdx.x;
  if (i < n) out[i] = f2bf(in[i]);
}

__global__ void zero_f32_kernel(float* __restrict__ p, int n) {
  int i = blockIdx.x * blockDim.x + threadIdx.x;
  if (i < n) p[i] = 0.0f;
}

// ---------------------------------------------------------------------------
// LSTM encoder: 16 sequences / block, Whh+Wih bf16 resident in LDS,
// per-step gates = [xt | h] WMMA GEMMs, enc_fc epilogue via WMMA.
// Sequences: 0..127 = ego, 128..4223 = agents (b*32+a).
// ---------------------------------------------------------------------------
#define LSTM_SMEM_BYTES 107072

__global__ __launch_bounds__(BDIM) void lstm_kernel(
    const float* __restrict__ ego_state, const float* __restrict__ agents_state,
    const int* __restrict__ ego_len, const int* __restrict__ ag_len,
    const float* __restrict__ sfc_W, const float* __restrict__ sfc_b,
    const float* __restrict__ Wih, const float* __restrict__ Whh,
    const float* __restrict__ bih, const float* __restrict__ bhh,
    const float* __restrict__ enc_W, const float* __restrict__ enc_b,
    float* __restrict__ enc_out) {
  extern __shared__ unsigned char smem[];
  float* gates = (float*)smem;                 // 16*256
  float* cst   = gates + 4096;                 // 16*64
  float* last  = cst + 1024;                   // 16*64
  float* seqs  = last + 1024;                  // 16*32*5
  float* bsum  = seqs + 2560;                  // 256
  float* sfw   = bsum + 256;                   // 64*5
  float* sfb   = sfw + 320;                    // 64
  int*   capT  = (int*)(sfb + 64);             // 16
  unsigned short* WihS = (unsigned short*)(capT + 16);  // 256*64 bf16
  unsigned short* WhhS = WihS + 16384;                  // 256*64 bf16
  unsigned short* xtS  = WhhS + 16384;                  // 16*64 bf16
  unsigned short* hbS  = xtS + 1024;                    // 16*64 bf16

  int tid = threadIdx.x;
  int wave = tid >> 5;
  int s0 = blockIdx.x * 16;

  for (int i = tid; i < 16384; i += BDIM) {
    WihS[i] = f2bf(Wih[i]);
    WhhS[i] = f2bf(Whh[i]);
  }
  if (tid < 256) bsum[tid] = bih[tid] + bhh[tid];
  if (tid < 64) sfb[tid] = sfc_b[tid];
  for (int i = tid; i < 320; i += BDIM) sfw[i] = sfc_W[i];
  for (int i = tid; i < 2560; i += BDIM) {
    int si = i / 160, e = i % 160;
    int s = s0 + si;
    seqs[i] = (s < 128) ? ego_state[s * 160 + e]
                        : agents_state[(size_t)(s - 128) * 160 + e];
  }
  if (tid < 16) {
    int s = s0 + tid;
    int L = (s < 128) ? ego_len[s] : ag_len[s - 128];
    L = L < 1 ? 1 : (L > 32 ? 32 : L);
    capT[tid] = L - 1;
  }
  {
    int i = tid & 15, jg = tid >> 4;
#pragma unroll
    for (int jj = 0; jj < 4; ++jj) {
      int j = jg * 4 + jj;
      cst[i * 64 + j] = 0.0f;
      last[i * 64 + j] = 0.0f;
      hbS[i * 64 + j] = 0;
    }
  }
  __syncthreads();

  int nt0 = wave * 2, nt1 = nt0 + 1;
  for (int t = 0; t < 32; ++t) {
    {  // xt = state_fc(seq[:, t, :])   (K=5 -> VALU)
      int i = tid & 15, jg = tid >> 4;
      const float* sp = seqs + (i * 32 + t) * 5;
#pragma unroll
      for (int jj = 0; jj < 4; ++jj) {
        int j = jg * 4 + jj;
        float acc = sfb[j];
#pragma unroll
        for (int k = 0; k < 5; ++k) acc += sp[k] * sfw[j * 5 + k];
        xtS[i * 64 + j] = f2bf(acc);
      }
    }
    __syncthreads();
    // gates[16x256] = xt @ Wih^T + h @ Whh^T  (WMMA, 8 per wave per step)
    v16bf ax0 = frag_ld(xtS, 64, 0);
    v16bf ax1 = frag_ld(xtS, 64, 32);
    v16bf ah0 = frag_ld(hbS, 64, 0);
    v16bf ah1 = frag_ld(hbS, 64, 32);
    v8f g0 = {}; v8f g1 = {};
    {
      const unsigned short* Bi = WihS + nt0 * 1024;
      const unsigned short* Bh = WhhS + nt0 * 1024;
      g0 = wmma_bf16(ax0, frag_ld(Bi, 64, 0), g0);
      g0 = wmma_bf16(ax1, frag_ld(Bi, 64, 32), g0);
      g0 = wmma_bf16(ah0, frag_ld(Bh, 64, 0), g0);
      g0 = wmma_bf16(ah1, frag_ld(Bh, 64, 32), g0);
    }
    {
      const unsigned short* Bi = WihS + nt1 * 1024;
      const unsigned short* Bh = WhhS + nt1 * 1024;
      g1 = wmma_bf16(ax0, frag_ld(Bi, 64, 0), g1);
      g1 = wmma_bf16(ax1, frag_ld(Bi, 64, 32), g1);
      g1 = wmma_bf16(ah0, frag_ld(Bh, 64, 0), g1);
      g1 = wmma_bf16(ah1, frag_ld(Bh, 64, 32), g1);
    }
    store_tile(gates, 256, 0, nt0 * 16, 16, 256, g0, nullptr, 0);
    store_tile(gates, 256, 0, nt1 * 16, 16, 256, g1, nullptr, 0);
    __syncthreads();
    {  // elementwise LSTM cell update
      int i = tid & 15, jg = tid >> 4;
      int ct = capT[i];
#pragma unroll
      for (int jj = 0; jj < 4; ++jj) {
        int j = jg * 4 + jj;
        float ig = sigmf(gates[i * 256 + j] + bsum[j]);
        float fg = sigmf(gates[i * 256 + 64 + j] + bsum[64 + j]);
        float gg = tanhf(gates[i * 256 + 128 + j] + bsum[128 + j]);
        float og = sigmf(gates[i * 256 + 192 + j] + bsum[192 + j]);
        float c = fg * cst[i * 64 + j] + ig * gg;
        float h = og * tanhf(c);
        cst[i * 64 + j] = c;
        hbS[i * 64 + j] = f2bf(h);
        if (t == ct) last[i * 64 + j] = h;
      }
    }
    __syncthreads();
  }

  // enc = last @ enc_W^T + enc_b    (reuse WihS for enc_W bf16)
  for (int i = tid; i < 4096; i += BDIM) WihS[i] = f2bf(enc_W[i]);
  {
    int i = tid & 15, jg = tid >> 4;
#pragma unroll
    for (int jj = 0; jj < 4; ++jj) {
      int j = jg * 4 + jj;
      xtS[i * 64 + j] = f2bf(last[i * 64 + j]);
    }
  }
  __syncthreads();
  if (wave < 4) {
    int nt = wave;
    v16bf a0 = frag_ld(xtS, 64, 0);
    v16bf a1 = frag_ld(xtS, 64, 32);
    v8f e = {};
    const unsigned short* Bw = WihS + nt * 1024;
    e = wmma_bf16(a0, frag_ld(Bw, 64, 0), e);
    e = wmma_bf16(a1, frag_ld(Bw, 64, 32), e);
    int lane = tid & 31;
    int n = nt * 16 + (lane & 15);
    int mb = (lane & 16) ? 8 : 0;
    float bv = enc_b[n];
#pragma unroll
    for (int j = 0; j < 8; ++j) {
      int m = mb + j;
      enc_out[(size_t)(s0 + m) * 64 + n] = e[j] + bv;
    }
  }
}

// ---------------------------------------------------------------------------
// Scatter-add agent encodings into 14x14x64 grid (overlaps accumulate).
// ---------------------------------------------------------------------------
__global__ void scatter_kernel(const float* __restrict__ enc,
                               const int* __restrict__ pos,
                               float* __restrict__ grid) {
  int idx = blockIdx.x * blockDim.x + threadIdx.x;
  if (idx >= 128 * 32 * 64) return;
  int j = idx & 63;
  int ba = idx >> 6;  // b*32 + a
  int b = ba >> 5;
  int y = pos[ba * 2 + 0];
  int x = pos[ba * 2 + 1];
  float v = enc[(size_t)(128 + ba) * 64 + j];
  atomicAdd(&grid[((size_t)b * 196 + y * 14 + x) * 64 + j], v);
}

// ---------------------------------------------------------------------------
// Pack [cnn_feat | grid] into padded bf16 [B][1088][16][16] (zero halo).
// ---------------------------------------------------------------------------
__global__ void pack1_kernel(const float* __restrict__ cnn,
                             const float* __restrict__ grid,
                             unsigned short* __restrict__ xpad) {
  int idx = blockIdx.x * blockDim.x + threadIdx.x;
  if (idx >= 128 * 1088 * 256) return;
  int px = idx & 15;
  int py = (idx >> 4) & 15;
  int rem = idx >> 8;
  int ch = rem % 1088;
  int b = rem / 1088;
  float v = 0.0f;
  if (py >= 1 && py <= 14 && px >= 1 && px <= 14) {
    int iy = py - 1, ix = px - 1;
    v = (ch < 1024) ? cnn[(((size_t)b * 1024 + ch) * 14 + iy) * 14 + ix]
                    : grid[((size_t)b * 196 + iy * 14 + ix) * 64 + (ch - 1024)];
  }
  xpad[idx] = f2bf(v);
}

// ---------------------------------------------------------------------------
// Implicit-GEMM 3x3 conv, bf16 WMMA.  out[B][Cout][P] f32.
// xpad: [B][Cin][16][16] bf16 (pad=1 halo included). stride 1 or 2.
// Block tile: 128 out-channels x 64 pixels; wave = 32x32 quadrant,
// 2 A-frags + 2 B-frags -> 4 WMMAs per K chunk.
// ---------------------------------------------------------------------------
__global__ __launch_bounds__(BDIM) void conv3x3_wmma_kernel(
    const unsigned short* __restrict__ xpad, const unsigned short* __restrict__ wbf,
    float* __restrict__ out, int Cin, int Cout, int outW, int P, int stride) {
  __shared__ unsigned short As[128 * 34];
  __shared__ unsigned short Bs[64 * 34];
  int b = blockIdx.x;
  int m0 = blockIdx.y * 128;
  int p0 = blockIdx.z * 64;
  int tid = threadIdx.x;
  int wave = tid >> 5;
  int wm = wave >> 1;  // 0..3 : 32-row band
  int wn = wave & 1;   // 0..1 : 32-col band
  v8f acc00 = {}; v8f acc01 = {}; v8f acc10 = {}; v8f acc11 = {};
  const unsigned short* xb = xpad + (size_t)b * Cin * 256;
  for (int tap = 0; tap < 9; ++tap) {
    int r = tap / 3, s = tap % 3;
    for (int c0 = 0; c0 < Cin; c0 += 32) {
      __syncthreads();
      for (int i = tid; i < 6144; i += BDIM) {
        if (i < 4096) {
          int row = i >> 5, cc = i & 31;
          int m = m0 + row, ch = c0 + cc;
          As[row * 34 + cc] = wbf[((size_t)m * Cin + ch) * 9 + tap];
        } else {
          int j = i - 4096;
          int row = j >> 5, cc = j & 31;
          int p = p0 + row, ch = c0 + cc;
          unsigned short xv = 0;
          if (p < P) {
            int py = (p / outW) * stride + r;
            int px = (p % outW) * stride + s;
            xv = xb[((size_t)ch * 16 + py) * 16 + px];
          }
          Bs[row * 34 + cc] = xv;
        }
      }
      __syncthreads();
      v16bf a0 = frag_ld(As + (wm * 2 + 0) * 16 * 34, 34, 0);
      v16bf a1 = frag_ld(As + (wm * 2 + 1) * 16 * 34, 34, 0);
      v16bf b0 = frag_ld(Bs + (wn * 2 + 0) * 16 * 34, 34, 0);
      v16bf b1 = frag_ld(Bs + (wn * 2 + 1) * 16 * 34, 34, 0);
      acc00 = wmma_bf16(a0, b0, acc00);
      acc01 = wmma_bf16(a0, b1, acc01);
      acc10 = wmma_bf16(a1, b0, acc10);
      acc11 = wmma_bf16(a1, b1, acc11);
    }
  }
  float* outb = out + (size_t)b * Cout * P;
  int ms = m0 + wm * 32, ns = p0 + wn * 32;
  store_tile(outb, P, ms + 0,  ns + 0,  Cout, P, acc00, nullptr, 0);
  store_tile(outb, P, ms + 0,  ns + 16, Cout, P, acc01, nullptr, 0);
  store_tile(outb, P, ms + 16, ns + 0,  Cout, P, acc10, nullptr, 0);
  store_tile(outb, P, ms + 16, ns + 16, Cout, P, acc11, nullptr, 0);
}

// ---------------------------------------------------------------------------
// BatchNorm statistics -> scale/shift per channel.
// ---------------------------------------------------------------------------
__global__ __launch_bounds__(BDIM) void bn_stats_kernel(
    const float* __restrict__ x, int C, int P, int NB,
    const float* __restrict__ g, const float* __restrict__ bta,
    float* __restrict__ scale, float* __restrict__ shift) {
  __shared__ float s1[BDIM];
  __shared__ float s2[BDIM];
  int ch = blockIdx.x;
  int tid = threadIdx.x;
  float sum = 0.0f, sq = 0.0f;
  int n = NB * P;
  for (int i = tid; i < n; i += BDIM) {
    int bb = i / P, p = i % P;
    float v = x[((size_t)bb * C + ch) * P + p];
    sum += v;
    sq += v * v;
  }
  s1[tid] = sum; s2[tid] = sq;
  __syncthreads();
  for (int off = BDIM / 2; off > 0; off >>= 1) {
    if (tid < off) { s1[tid] += s1[tid + off]; s2[tid] += s2[tid + off]; }
    __syncthreads();
  }
  if (tid == 0) {
    float mean = s1[0] / (float)n;
    float var = s2[0] / (float)n - mean * mean;
    float sc = g[ch] * rsqrtf(var + 1e-5f);
    scale[ch] = sc;
    shift[ch] = bta[ch] - mean * sc;
  }
}

// BN+ReLU on conv1 output, repacked padded bf16 [B][512][16][16].
__global__ void bnrelu_pack_kernel(const float* __restrict__ c1,
                                   const float* __restrict__ scale,
                                   const float* __restrict__ shift,
                                   unsigned short* __restrict__ x2pad) {
  int idx = blockIdx.x * blockDim.x + threadIdx.x;
  if (idx >= 128 * 512 * 256) return;
  int px = idx & 15;
  int py = (idx >> 4) & 15;
  int rem = idx >> 8;
  int ch = rem % 512;
  int b = rem / 512;
  float v = 0.0f;
  if (py >= 1 && py <= 14 && px >= 1 && px <= 14) {
    float u = c1[((size_t)b * 512 + ch) * 196 + (py - 1) * 14 + (px - 1)];
    v = fmaxf(u * scale[ch] + shift[ch], 0.0f);
  }
  x2pad[idx] = f2bf(v);
}

// BN+ReLU on conv2 output, pixel-major bf16 [B*49][512] for 1x1-conv GEMMs.
__global__ void bnrelu2_kernel(const float* __restrict__ c2,
                               const float* __restrict__ scale,
                               const float* __restrict__ shift,
                               unsigned short* __restrict__ x2bf) {
  int idx = blockIdx.x * blockDim.x + threadIdx.x;
  if (idx >= 128 * 512 * 49) return;
  int p = idx % 49;
  int rem = idx / 49;
  int ch = rem % 512;
  int b = rem / 512;
  float u = c2[((size_t)b * 512 + ch) * 49 + p];
  float v = fmaxf(u * scale[ch] + shift[ch], 0.0f);
  x2bf[((size_t)(b * 49 + p)) * 512 + ch] = f2bf(v);
}

// ---------------------------------------------------------------------------
// Generic bf16 WMMA GEMM: C[M,N] = act(A[M,K] * W[N,K]^T + bias)
// Block tile 128x64; wave = 32x32 quadrant, 4 WMMAs per K chunk.
// ---------------------------------------------------------------------------
__global__ __launch_bounds__(BDIM) void gemm_bf16_kernel(
    const unsigned short* __restrict__ A, int lda,
    const unsigned short* __restrict__ W, int ldb, float* __restrict__ C,
    int ldc, const float* __restrict__ bias, int M, int N, int K, int relu) {
  __shared__ unsigned short As[128 * 34];
  __shared__ unsigned short Ws[64 * 34];
  int tid = threadIdx.x;
  int wave = tid >> 5;
  int m0 = blockIdx.x * 128, n0 = blockIdx.y * 64;
  int wm = wave >> 1;  // 0..3
  int wn = wave & 1;   // 0..1
  v8f acc00 = {}; v8f acc01 = {}; v8f acc10 = {}; v8f acc11 = {};
  for (int k0 = 0; k0 < K; k0 += 32) {
    __syncthreads();
    for (int i = tid; i < 6144; i += BDIM) {
      if (i < 4096) {
        int row = i >> 5, cc = i & 31;
        int gm = m0 + row, gk = k0 + cc;
        As[row * 34 + cc] =
            (gm < M && gk < K) ? A[(size_t)gm * lda + gk] : (unsigned short)0;
      } else {
        int j = i - 4096;
        int row = j >> 5, cc = j & 31;
        int gn = n0 + row, gk = k0 + cc;
        Ws[row * 34 + cc] =
            (gn < N && gk < K) ? W[(size_t)gn * ldb + gk] : (unsigned short)0;
      }
    }
    if (k0 + 32 < K) {  // hint next K chunk toward the caches
      int pr = m0 + (tid >> 1); if (pr >= M) pr = M - 1;
      int pn = n0 + (tid & 63); if (pn >= N) pn = N - 1;
      __builtin_prefetch(A + (size_t)pr * lda + (k0 + 32), 0, 0);
      __builtin_prefetch(W + (size_t)pn * ldb + (k0 + 32), 0, 0);
    }
    __syncthreads();
    v16bf a0 = frag_ld(As + (wm * 2 + 0) * 16 * 34, 34, 0);
    v16bf a1 = frag_ld(As + (wm * 2 + 1) * 16 * 34, 34, 0);
    v16bf b0 = frag_ld(Ws + (wn * 2 + 0) * 16 * 34, 34, 0);
    v16bf b1 = frag_ld(Ws + (wn * 2 + 1) * 16 * 34, 34, 0);
    acc00 = wmma_bf16(a0, b0, acc00);
    acc01 = wmma_bf16(a0, b1, acc01);
    acc10 = wmma_bf16(a1, b0, acc10);
    acc11 = wmma_bf16(a1, b1, acc11);
  }
  int ms = m0 + wm * 32, ns = n0 + wn * 32;
  store_tile(C, ldc, ms + 0,  ns + 0,  M, N, acc00, bias, relu);
  store_tile(C, ldc, ms + 0,  ns + 16, M, N, acc01, bias, relu);
  store_tile(C, ldc, ms + 16, ns + 0,  M, N, acc10, bias, relu);
  store_tile(C, ldc, ms + 16, ns + 16, M, N, acc11, bias, relu);
}

// ---------------------------------------------------------------------------
// ego_t = [x[:,:,3,3] (512) | ego_enc (64)] -> f32 + bf16 copies
// ---------------------------------------------------------------------------
__global__ void egot_kernel(const unsigned short* __restrict__ x2bf,
                            const float* __restrict__ enc,
                            float* __restrict__ egot,
                            unsigned short* __restrict__ egotbf) {
  int idx = blockIdx.x * blockDim.x + threadIdx.x;
  if (idx >= 128 * 576) return;
  int b = idx / 576, j = idx % 576;
  float v = (j < 512) ? bf2f(x2bf[(size_t)(b * 49 + 24) * 512 + j])
                      : enc[b * 64 + (j - 512)];
  egot[idx] = v;
  egotbf[idx] = f2bf(v);
}

// ---------------------------------------------------------------------------
// Attention over 9x9 grid (pad-1 1x1 conv => border k,v are exactly zero, but
// those zero energies still participate in the softmax normalization).
// One block per (b, head).
// ---------------------------------------------------------------------------
__global__ __launch_bounds__(128) void attn_kernel(
    const float* __restrict__ qmat, const float* __restrict__ kmat,
    const float* __restrict__ vmat, float* __restrict__ omat) {
  __shared__ float qs[64];
  __shared__ float es[81];
  __shared__ float invsum;
  int b = blockIdx.x >> 4, h = blockIdx.x & 15;
  int tid = threadIdx.x;
  if (tid < 64) qs[tid] = qmat[(size_t)b * 1024 + h * 64 + tid];
  __syncthreads();
  if (tid < 81) {
    int gy = tid / 9, gx = tid % 9;
    float e = 0.0f;
    if (gy >= 1 && gy <= 7 && gx >= 1 && gx <= 7) {
      int p = (gy - 1) * 7 + (gx - 1);
      const float* kp = kmat + ((size_t)(b * 49 + p) * 1024 + h * 64);
      for (int d = 0; d < 64; ++d) e += qs[d] * kp[d];
    }
    es[tid] = e * (1.0f / 32.0f);  // / sqrt(NHEAD*DHEAD)
  }
  __syncthreads();
  if (tid == 0) {
    float mx = es[0];
    for (int i = 1; i < 81; ++i) mx = fmaxf(mx, es[i]);
    float sm = 0.0f;
    for (int i = 0; i < 81; ++i) { es[i] = expf(es[i] - mx); sm += es[i]; }
    invsum = 1.0f / sm;
  }
  __syncthreads();
  if (tid < 64) {
    float acc = 0.0f;
    for (int gy = 1; gy <= 7; ++gy)
      for (int gx = 1; gx <= 7; ++gx) {
        int p = (gy - 1) * 7 + (gx - 1);
        acc += es[gy * 9 + gx] *
               vmat[(size_t)(b * 49 + p) * 1024 + h * 64 + tid];
      }
    omat[(size_t)b * 1024 + h * 64 + tid] = acc * invsum;
  }
}

// feat = [attn_out (512) | ego_t (576)] -> bf16
__global__ void featbf_kernel(const float* __restrict__ aout,
                              const float* __restrict__ egot,
                              unsigned short* __restrict__ featbf) {
  int idx = blockIdx.x * blockDim.x + threadIdx.x;
  if (idx >= 128 * 1088) return;
  int b = idx / 1088, j = idx % 1088;
  float v = (j < 512) ? aout[b * 512 + j] : egot[b * 576 + (j - 512)];
  featbf[idx] = f2bf(v);
}

// ---------------------------------------------------------------------------
// Normalized Legendre basis tmat[8][19] (single thread; tiny).
// ---------------------------------------------------------------------------
__global__ void tmat_kernel(float* __restrict__ tmat) {
  if (threadIdx.x == 0 && blockIdx.x == 0) {
    float t[19];
    float P[8][19];
    for (int i = 0; i < 19; ++i) t[i] = (float)(i - 6) / 12.0f;
    for (int i = 0; i < 19; ++i) { P[0][i] = 1.0f; P[1][i] = t[i]; }
    for (int n = 1; n < 7; ++n)
      for (int i = 0; i < 19; ++i)
        P[n + 1][i] =
            ((float)(2 * n + 1) * t[i] * P[n][i] - (float)n * P[n - 1][i]) /
            (float)(n + 1);
    for (int n = 0; n < 8; ++n) {
      float norm = sqrtf((2.0f * n + 1.0f) * 0.5f);
      for (int i = 0; i < 19; ++i) tmat[n * 19 + i] = P[n][i] * norm;
    }
  }
}

// traj[b][m][t][{x,y}] = coeffs . tmat
__global__ void traj_kernel(const float* __restrict__ d2,
                            const float* __restrict__ tmat,
                            float* __restrict__ traj) {
  int idx = blockIdx.x * blockDim.x + threadIdx.x;
  if (idx >= 128 * 25 * 19) return;
  int tt = idx % 19;
  int rem = idx / 19;
  int m = rem % 25;
  int b = rem / 25;
  const float* c = d2 + ((size_t)b * 400 + m * 16);
  float x = 0.0f, y = 0.0f;
#pragma unroll
  for (int n = 0; n < 8; ++n) {
    float tv = tmat[n * 19 + tt];
    x += c[n] * tv;
    y += c[8 + n] * tv;
  }
  traj[(size_t)idx * 2 + 0] = x;
  traj[(size_t)idx * 2 + 1] = y;
}

// ---------------------------------------------------------------------------
// Host-side orchestration.
// ---------------------------------------------------------------------------
extern "C" void kernel_launch(void* const* d_in, const int* in_sizes, int n_in,
                              void* d_out, int out_size, void* d_ws,
                              size_t ws_size, hipStream_t stream) {
  (void)in_sizes; (void)n_in; (void)out_size; (void)ws_size;

  const float* cnn_feat     = (const float*)d_in[0];
  const float* ego_state    = (const float*)d_in[1];
  const float* agents_state = (const float*)d_in[2];
  const int*   ego_len      = (const int*)d_in[3];
  const int*   ag_len       = (const int*)d_in[4];
  const int*   ag_pos       = (const int*)d_in[5];
  const float* sfc_W        = (const float*)d_in[6];
  const float* sfc_b        = (const float*)d_in[7];
  const float* lstm_Wih     = (const float*)d_in[8];
  const float* lstm_Whh     = (const float*)d_in[9];
  const float* lstm_bih     = (const float*)d_in[10];
  const float* lstm_bhh     = (const float*)d_in[11];
  const float* enc_fc_W     = (const float*)d_in[12];
  const float* enc_fc_b     = (const float*)d_in[13];
  const float* conv1_W      = (const float*)d_in[14];
  const float* bn1_g        = (const float*)d_in[15];
  const float* bn1_b        = (const float*)d_in[16];
  const float* conv2_W      = (const float*)d_in[17];
  const float* bn2_g        = (const float*)d_in[18];
  const float* bn2_b        = (const float*)d_in[19];
  const float* q_W          = (const float*)d_in[20];
  const float* k_W          = (const float*)d_in[21];
  const float* v_W          = (const float*)d_in[22];
  const float* ao_W         = (const float*)d_in[23];
  const float* ao_b         = (const float*)d_in[24];
  const float* cls_W        = (const float*)d_in[25];
  const float* cls_b        = (const float*)d_in[26];
  const float* dec1_W       = (const float*)d_in[27];
  const float* dec1_b       = (const float*)d_in[28];
  const float* dec2_W       = (const float*)d_in[29];
  const float* dec2_b       = (const float*)d_in[30];

  float* out_traj = (float*)d_out;                 // [128][25][19][2]
  float* out_conf = out_traj + 128 * 25 * 19 * 2;  // [128][25]

  // ---- workspace bump allocator (256B aligned) ----
  char* ws = (char*)d_ws;
  size_t off = 0;
  auto alloc = [&](size_t bytes) -> char* {
    char* p = ws + off;
    off += (bytes + 255) & ~((size_t)255);
    return p;
  };
  unsigned short* w1bf   = (unsigned short*)alloc((size_t)512 * 1088 * 9 * 2);
  unsigned short* w2bf   = (unsigned short*)alloc((size_t)512 * 512 * 9 * 2);
  unsigned short* wqbf   = (unsigned short*)alloc((size_t)1024 * 576 * 2);
  unsigned short* wkbf   = (unsigned short*)alloc((size_t)1024 * 512 * 2);
  unsigned short* wvbf   = (unsigned short*)alloc((size_t)1024 * 512 * 2);
  unsigned short* wobf   = (unsigned short*)alloc((size_t)512 * 1024 * 2);
  unsigned short* wclsbf = (unsigned short*)alloc((size_t)25 * 1088 * 2);
  unsigned short* wd1bf  = (unsigned short*)alloc((size_t)256 * 1088 * 2);
  unsigned short* wd2bf  = (unsigned short*)alloc((size_t)400 * 256 * 2);
  unsigned short* xpad   = (unsigned short*)alloc((size_t)128 * 1088 * 256 * 2);
  float* grid            = (float*)alloc((size_t)128 * 196 * 64 * 4);
  float* enc             = (float*)alloc((size_t)4224 * 64 * 4);
  float* c1out           = (float*)alloc((size_t)128 * 512 * 196 * 4);
  float* sc1             = (float*)alloc(512 * 4);
  float* sh1             = (float*)alloc(512 * 4);
  unsigned short* x2pad  = (unsigned short*)alloc((size_t)128 * 512 * 256 * 2);
  float* c2out           = (float*)alloc((size_t)128 * 512 * 49 * 4);
  float* sc2             = (float*)alloc(512 * 4);
  float* sh2             = (float*)alloc(512 * 4);
  unsigned short* x2bf   = (unsigned short*)alloc((size_t)6272 * 512 * 2);
  float* egot            = (float*)alloc((size_t)128 * 576 * 4);
  unsigned short* egotbf = (unsigned short*)alloc((size_t)128 * 576 * 2);
  float* qmat            = (float*)alloc((size_t)128 * 1024 * 4);
  float* kmat            = (float*)alloc((size_t)6272 * 1024 * 4);
  float* vmat            = (float*)alloc((size_t)6272 * 1024 * 4);
  float* omat            = (float*)alloc((size_t)128 * 1024 * 4);
  unsigned short* obf    = (unsigned short*)alloc((size_t)128 * 1024 * 2);
  float* aout            = (float*)alloc((size_t)128 * 512 * 4);
  unsigned short* featbf = (unsigned short*)alloc((size_t)128 * 1088 * 2);
  float* d1              = (float*)alloc((size_t)128 * 256 * 4);
  unsigned short* d1bf   = (unsigned short*)alloc((size_t)128 * 256 * 2);
  float* d2              = (float*)alloc((size_t)128 * 400 * 4);
  float* tmat            = (float*)alloc(8 * 19 * 4);

  auto blocks = [](int n) { return (n + BDIM - 1) / BDIM; };

  // ---- weight conversions + tmat ----
  tmat_kernel<<<1, 32, 0, stream>>>(tmat);
  f2bf_kernel<<<blocks(512 * 1088 * 9), BDIM, 0, stream>>>(conv1_W, w1bf, 512 * 1088 * 9);
  f2bf_kernel<<<blocks(512 * 512 * 9), BDIM, 0, stream>>>(conv2_W, w2bf, 512 * 512 * 9);
  f2bf_kernel<<<blocks(1024 * 576), BDIM, 0, stream>>>(q_W, wqbf, 1024 * 576);
  f2bf_kernel<<<blocks(1024 * 512), BDIM, 0, stream>>>(k_W, wkbf, 1024 * 512);
  f2bf_kernel<<<blocks(1024 * 512), BDIM, 0, stream>>>(v_W, wvbf, 1024 * 512);
  f2bf_kernel<<<blocks(512 * 1024), BDIM, 0, stream>>>(ao_W, wobf, 512 * 1024);
  f2bf_kernel<<<blocks(25 * 1088), BDIM, 0, stream>>>(cls_W, wclsbf, 25 * 1088);
  f2bf_kernel<<<blocks(256 * 1088), BDIM, 0, stream>>>(dec1_W, wd1bf, 256 * 1088);
  f2bf_kernel<<<blocks(400 * 256), BDIM, 0, stream>>>(dec2_W, wd2bf, 400 * 256);

  // ---- LSTM encoders (ego + agents fused, 16 seqs/block) ----
  lstm_kernel<<<264, BDIM, LSTM_SMEM_BYTES, stream>>>(
      ego_state, agents_state, ego_len, ag_len, sfc_W, sfc_b, lstm_Wih,
      lstm_Whh, lstm_bih, lstm_bhh, enc_fc_W, enc_fc_b, enc);

  // ---- agent scatter into spatial grid ----
  zero_f32_kernel<<<blocks(128 * 196 * 64), BDIM, 0, stream>>>(grid, 128 * 196 * 64);
  scatter_kernel<<<blocks(128 * 32 * 64), BDIM, 0, stream>>>(enc, ag_pos, grid);

  // ---- conv1 (1088 -> 512, 14x14) ----
  pack1_kernel<<<blocks(128 * 1088 * 256), BDIM, 0, stream>>>(cnn_feat, grid, xpad);
  conv3x3_wmma_kernel<<<dim3(128, 4, 4), BDIM, 0, stream>>>(
      xpad, w1bf, c1out, 1088, 512, 14, 196, 1);
  bn_stats_kernel<<<512, BDIM, 0, stream>>>(c1out, 512, 196, 128, bn1_g, bn1_b, sc1, sh1);
  bnrelu_pack_kernel<<<blocks(128 * 512 * 256), BDIM, 0, stream>>>(c1out, sc1, sh1, x2pad);

  // ---- conv2 (512 -> 512, stride 2, 7x7) ----
  conv3x3_wmma_kernel<<<dim3(128, 4, 1), BDIM, 0, stream>>>(
      x2pad, w2bf, c2out, 512, 512, 7, 49, 2);
  bn_stats_kernel<<<512, BDIM, 0, stream>>>(c2out, 512, 49, 128, bn2_g, bn2_b, sc2, sh2);
  bnrelu2_kernel<<<blocks(128 * 512 * 49), BDIM, 0, stream>>>(c2out, sc2, sh2, x2bf);

  // ---- attention projections (all via generic WMMA GEMM) ----
  egot_kernel<<<blocks(128 * 576), BDIM, 0, stream>>>(x2bf, enc, egot, egotbf);
  gemm_bf16_kernel<<<dim3(1, 16), BDIM, 0, stream>>>(
      egotbf, 576, wqbf, 576, qmat, 1024, nullptr, 128, 1024, 576, 0);
  gemm_bf16_kernel<<<dim3(49, 16), BDIM, 0, stream>>>(
      x2bf, 512, wkbf, 512, kmat, 1024, nullptr, 6272, 1024, 512, 0);
  gemm_bf16_kernel<<<dim3(49, 16), BDIM, 0, stream>>>(
      x2bf, 512, wvbf, 512, vmat, 1024, nullptr, 6272, 1024, 512, 0);
  attn_kernel<<<128 * 16, 128, 0, stream>>>(qmat, kmat, vmat, omat);
  f2bf_kernel<<<blocks(128 * 1024), BDIM, 0, stream>>>(omat, obf, 128 * 1024);
  gemm_bf16_kernel<<<dim3(1, 8), BDIM, 0, stream>>>(
      obf, 1024, wobf, 1024, aout, 512, ao_b, 128, 512, 1024, 0);

  // ---- heads ----
  featbf_kernel<<<blocks(128 * 1088), BDIM, 0, stream>>>(aout, egot, featbf);
  gemm_bf16_kernel<<<dim3(1, 1), BDIM, 0, stream>>>(
      featbf, 1088, wclsbf, 1088, out_conf, 25, cls_b, 128, 25, 1088, 0);
  gemm_bf16_kernel<<<dim3(1, 4), BDIM, 0, stream>>>(
      featbf, 1088, wd1bf, 1088, d1, 256, dec1_b, 128, 256, 1088, 1);
  f2bf_kernel<<<blocks(128 * 256), BDIM, 0, stream>>>(d1, d1bf, 128 * 256);
  gemm_bf16_kernel<<<dim3(1, 7), BDIM, 0, stream>>>(
      d1bf, 256, wd2bf, 256, d2, 400, dec2_b, 128, 400, 256, 0);
  traj_kernel<<<blocks(128 * 25 * 19), BDIM, 0, stream>>>(d2, tmat, out_traj);
}
